// LSTM_MODEL_80814104641853
// MI455X (gfx1250) — compile-verified, matrix operand order checked by reading, not run on hardware
//
#include <hip/hip_runtime.h>
#include <hip/hip_bf16.h>

// Problem dims (fixed by reference)
#define BATCH 256
#define SEQT  512
#define INP   128
#define HID   256
#define GAT   1024   // 4*HID

typedef __attribute__((ext_vector_type(16))) __bf16 v16bf;
typedef __attribute__((ext_vector_type(8)))  float  v8f;

union Frag16 { v16bf v; uint4 q[2]; unsigned short s[16]; };
union FragC  { v8f   v; float f[8]; };
union Pk8    { uint4 q; unsigned short s[8]; };

__device__ __forceinline__ unsigned short f2bf(float x){
  union { float f; unsigned u; } c; c.f = x;
  unsigned u = c.u;
  u += 0x7FFFu + ((u >> 16) & 1u);   // round-to-nearest-even
  return (unsigned short)(u >> 16);
}
__device__ __forceinline__ float bf2f(unsigned short b){
  union { unsigned u; float f; } c; c.u = ((unsigned)b) << 16;
  return c.f;
}

// ---------------------------------------------------------------------------
// Pack W (G x K, fp32) into WMMA B-fragment tiles (bf16).
// Tile (gt, kt) covers gate cols gt*16..+15, K rows kt*32..+31.
// Lane l (l<16): col n=l, K = {0..7} U {16..23}; lane l>=16: col n=l-16,
// K = {8..15} U {24..31}; 16 bf16 per lane, lane-contiguous in memory.
// ---------------------------------------------------------------------------
__global__ void pack_w(const float* __restrict__ W, unsigned short* __restrict__ out,
                       int G, int K){
  int tid = blockIdx.x * blockDim.x + threadIdx.x;     // one (tile,lane) per thread
  int KT = K >> 5;
  int ntl = (G >> 4) * KT * 32;
  if (tid >= ntl) return;
  int tile = tid >> 5, lane = tid & 31;
  int gt = tile / KT, kt = tile % KT;
  int n  = lane & 15;
  int g  = gt * 16 + n;
  int lo = (lane < 16) ? 0 : 8;
  unsigned short* dst = out + (size_t)tid * 16;
  #pragma unroll
  for (int j = 0; j < 16; ++j){
    int s = j >> 1, p = j & 1;
    int kk = (s < 4) ? (lo + 2 * s + p) : (16 + lo + 2 * (s - 4) + p);
    dst[j] = f2bf(W[(size_t)g * K + kt * 32 + kk]);
  }
}

__global__ void f32_to_bf16_k(const float* __restrict__ in,
                              unsigned short* __restrict__ out, size_t n){
  size_t i = (size_t)blockIdx.x * blockDim.x + threadIdx.x;
  if (i < n) out[i] = f2bf(in[i]);
}

__global__ void bias_sum(const float* __restrict__ a, const float* __restrict__ b,
                         float* __restrict__ o, int n){
  int i = blockIdx.x * blockDim.x + threadIdx.x;
  if (i < n) o[i] = a[i] + b[i];
}

// ---------------------------------------------------------------------------
// xW GEMM: gates[b][t][g] = sum_k Abf[b*T+t][k] * W[g][k] + bias[g]
// A rows are (b*T+t) row-major bf16; one wave per 16x16 output tile.
// Output is stored directly in the recurrence's C-fragment layout:
//   idx = (((t*16 + mt)*16 + nt)*4 + g4)*256 + lane*8 + r
// where mt=b>>4, nt=(g>>4)&15, g4=g>>8, lane=(g&15)+((b&8)?16:0), r=b&7,
// so the scan kernel reads one aligned b128 per gate per lane per step.
// ---------------------------------------------------------------------------
__global__ void gemm_xw(const unsigned short* __restrict__ Abf,
                        const unsigned short* __restrict__ Wp,
                        const float* __restrict__ bs,
                        unsigned short* __restrict__ Out,
                        int K){
  const int lane = threadIdx.x & 31;
  const int gw   = blockIdx.x * (blockDim.x >> 5) + (threadIdx.x >> 5);
  const int GT   = GAT / 16;            // 64 tile cols
  const int mtg  = gw / GT, gt = gw % GT;
  const int KT   = K >> 5;
  const int n    = lane & 15;
  const int half = (lane < 16) ? 0 : 8;

  FragC acc;
  const float bv = bs[gt * 16 + n];
  #pragma unroll
  for (int r = 0; r < 8; ++r) acc.f[r] = bv;

  const unsigned short* arow_p  = Abf + (size_t)(mtg * 16 + n) * K;
  const unsigned short* wp_tile = Wp + (size_t)gt * KT * 512;

  for (int kt = 0; kt < KT; ++kt){
    Frag16 a, b;
    const unsigned short* ap = arow_p + kt * 32 + half;
    a.q[0] = *(const uint4*)(ap);
    a.q[1] = *(const uint4*)(ap + 16);
    const unsigned short* bp = wp_tile + (size_t)kt * 512 + lane * 16;
    b.q[0] = *(const uint4*)(bp);
    b.q[1] = *(const uint4*)(bp + 8);
    acc.v = __builtin_amdgcn_wmma_f32_16x16x32_bf16(false, a.v, false, b.v,
                                                    (short)0, acc.v, false, false);
  }

  // scatter-store into scan-fragment layout (one u16 per r; b_ is tile-constant)
  const int brow0 = mtg * 16 + half;
  const int b_    = brow0 >> 9;                 // T = 512, tiles never cross b
  const int g4    = gt >> 4;
  const int nt    = gt & 15;
  const int mtc   = b_ >> 4;
  const int laneR = n + ((b_ & 8) ? 16 : 0);
  const int rR    = b_ & 7;
  #pragma unroll
  for (int r = 0; r < 8; ++r){
    int t_ = (brow0 + r) & (SEQT - 1);
    size_t idx = ((((size_t)t_ * 16 + mtc) * 16 + nt) * 4 + g4) * 256 + laneR * 8 + rR;
    Out[idx] = f2bf(acc.f[r]);
  }
}

// ---------------------------------------------------------------------------
// Persistent LSTM recurrence for one layer. 32 blocks x 256 threads.
// Block b: nt = b & 15 (hidden tile, shared by all 8 waves), waves cover
// mt = (b>>4)*8 + waveid. The block's 32 W_hh B-fragment tiles (32 KB) are
// staged in LDS once and served from ds_load_b128 every step. The cell
// state c lives in accumulator VGPRs for the whole scan; h is exchanged
// through global memory (bf16, double buffered), one grid barrier per step.
// ---------------------------------------------------------------------------
__global__ void lstm_recur(const unsigned short* __restrict__ XWp,  // packed gates
                           const unsigned short* __restrict__ Whhp, // packed GxH
                           unsigned short* __restrict__ hb0,
                           unsigned short* __restrict__ hb1,
                           unsigned short* __restrict__ hseq,       // [b*T+t][h] or null
                           float* __restrict__ hlast,               // [b][h]   or null
                           int* __restrict__ bar){                  // [0]=count [1]=sense
  __shared__ unsigned short bsm[32 * 512];       // 32 tiles x 1KB = 32 KB

  const int lane = threadIdx.x & 31;
  const int wid  = threadIdx.x >> 5;
  const int nt   = blockIdx.x & 15;
  const int mt   = (blockIdx.x >> 4) * 8 + wid;
  const int n    = lane & 15;
  const int half = (lane < 16) ? 0 : 8;
  const int KT   = HID / 32;             // 8
  const int nblocks = (int)gridDim.x;

  // ---- stage this block's W_hh fragments into LDS (invariant over scan) ----
  // tiles: tloc = g4*8 + kt  ->  global tile (g4*16 + nt, kt)
  for (int j = threadIdx.x; j < 32 * 64; j += blockDim.x){   // 64 uint4 per tile
    int tloc = j >> 6, qi = j & 63;
    int g4 = tloc >> 3, kt = tloc & 7;
    const uint4* src = (const uint4*)(Whhp + ((size_t)(g4 * 16 + nt) * KT + kt) * 512) + qi;
    *((uint4*)(bsm + (size_t)tloc * 512) + qi) = *src;
  }
  __syncthreads();

  FragC c;
  #pragma unroll
  for (int r = 0; r < 8; ++r) c.f[r] = 0.f;

  for (int t = 0; t < SEQT; ++t){
    const unsigned short* hprev = (t & 1) ? hb1 : hb0;
    unsigned short*       hnext = (t & 1) ? hb0 : hb1;

    // C-init from packed xW: one b128 per gate per lane
    FragC acc[4];
    {
      const unsigned short* xw =
          XWp + (((size_t)t * 16 + mt) * 16 + nt) * 4 * 256 + lane * 8;
      if (t + 1 < SEQT)
        __builtin_prefetch(xw + (size_t)16 * 16 * 4 * 256, 0, 1);
      #pragma unroll
      for (int g4 = 0; g4 < 4; ++g4){
        Pk8 p; p.q = *(const uint4*)(xw + (size_t)g4 * 256);
        #pragma unroll
        for (int r = 0; r < 8; ++r) acc[g4].f[r] = bf2f(p.s[r]);
      }
    }

    // gates += h_prev @ W_hh^T : A from global (L2), B from LDS
    const unsigned short* arow_p = hprev + (size_t)(mt * 16 + n) * HID;
    for (int kt = 0; kt < KT; ++kt){
      Frag16 a;
      const unsigned short* ap = arow_p + kt * 32 + half;
      a.q[0] = *(const uint4*)(ap);
      a.q[1] = *(const uint4*)(ap + 16);
      #pragma unroll
      for (int g4 = 0; g4 < 4; ++g4){
        Frag16 b;
        const unsigned short* bp = bsm + ((size_t)(g4 * 8 + kt) * 512) + lane * 16;
        b.q[0] = *(const uint4*)(bp);
        b.q[1] = *(const uint4*)(bp + 8);
        acc[g4].v = __builtin_amdgcn_wmma_f32_16x16x32_bf16(
            false, a.v, false, b.v, (short)0, acc[g4].v, false, false);
      }
    }

    // elementwise cell update; c stays in registers across all t
    #pragma unroll
    for (int r = 0; r < 8; ++r){
      float ig = 1.f / (1.f + __expf(-acc[0].f[r]));
      float fg = 1.f / (1.f + __expf(-acc[1].f[r]));
      float gg = tanhf(acc[2].f[r]);
      float og = 1.f / (1.f + __expf(-acc[3].f[r]));
      float cv = fg * c.f[r] + ig * gg;
      c.f[r] = cv;
      float hv = og * tanhf(cv);
      int bm = mt * 16 + half + r;
      int hi = nt * 16 + n;
      hnext[(size_t)bm * HID + hi] = f2bf(hv);
      if (hseq)  hseq[((size_t)bm * SEQT + t) * HID + hi] = f2bf(hv);
      if (hlast && t == SEQT - 1) hlast[(size_t)bm * HID + hi] = hv;
    }

    // grid-wide barrier (count + monotonic sense), polite spin
    __threadfence();
    __syncthreads();
    if (threadIdx.x == 0){
      if (__hip_atomic_fetch_add(&bar[0], 1, __ATOMIC_ACQ_REL,
                                 __HIP_MEMORY_SCOPE_AGENT) == nblocks - 1){
        __hip_atomic_store(&bar[0], 0, __ATOMIC_RELAXED, __HIP_MEMORY_SCOPE_AGENT);
        __hip_atomic_store(&bar[1], t + 1, __ATOMIC_RELEASE, __HIP_MEMORY_SCOPE_AGENT);
      } else {
        while (__hip_atomic_load(&bar[1], __ATOMIC_ACQUIRE,
                                 __HIP_MEMORY_SCOPE_AGENT) <= t)
          __builtin_amdgcn_s_sleep(1);
      }
    }
    __syncthreads();
  }
}

__global__ void fc_kernel(const float* __restrict__ hlast, const float* __restrict__ w,
                          const float* __restrict__ bias, float* __restrict__ out){
  int b = blockIdx.x * blockDim.x + threadIdx.x;
  if (b >= BATCH) return;
  float s = bias[0];
  #pragma unroll 4
  for (int h = 0; h < HID; ++h) s += hlast[(size_t)b * HID + h] * w[h];
  out[b] = s;
}

// ---------------------------------------------------------------------------
extern "C" void kernel_launch(void* const* d_in, const int* in_sizes, int n_in,
                              void* d_out, int out_size, void* d_ws, size_t ws_size,
                              hipStream_t stream){
  (void)in_sizes; (void)n_in; (void)out_size; (void)ws_size;
  const float* x     = (const float*)d_in[0];
  const float* Wih0  = (const float*)d_in[1];
  const float* Whh0  = (const float*)d_in[2];
  const float* bih0  = (const float*)d_in[3];
  const float* bhh0  = (const float*)d_in[4];
  const float* Wih1  = (const float*)d_in[5];
  const float* Whh1  = (const float*)d_in[6];
  const float* bih1  = (const float*)d_in[7];
  const float* bhh1  = (const float*)d_in[8];
  const float* fcw   = (const float*)d_in[9];
  const float* fcb   = (const float*)d_in[10];
  float* out = (float*)d_out;

  // workspace carve (256-byte aligned slabs)
  size_t off = 0;
  auto carve = [&](size_t bytes) -> char* {
    char* p = (char*)d_ws + off;
    off += (bytes + 255) & ~(size_t)255;
    return p;
  };
  unsigned short* x_bf  = (unsigned short*)carve((size_t)BATCH * SEQT * INP * 2);
  unsigned short* xw_bf = (unsigned short*)carve((size_t)BATCH * SEQT * GAT * 2);
  unsigned short* h1_bf = (unsigned short*)carve((size_t)BATCH * SEQT * HID * 2);
  unsigned short* wih0p = (unsigned short*)carve((size_t)GAT * INP * 2);
  unsigned short* whh0p = (unsigned short*)carve((size_t)GAT * HID * 2);
  unsigned short* wih1p = (unsigned short*)carve((size_t)GAT * HID * 2);
  unsigned short* whh1p = (unsigned short*)carve((size_t)GAT * HID * 2);
  float* bs0            = (float*)carve(GAT * 4);
  float* bs1            = (float*)carve(GAT * 4);
  unsigned short* hb0   = (unsigned short*)carve((size_t)BATCH * HID * 2);
  unsigned short* hb1   = (unsigned short*)carve((size_t)BATCH * HID * 2);
  float* hlast          = (float*)carve((size_t)BATCH * HID * 4);
  int* bar              = (int*)carve(256);

  // --- prep: pack weights to bf16 WMMA-B tiles, fold biases, convert x ---
  pack_w<<<(GAT/16 * INP/32 * 32 + 255) / 256, 256, 0, stream>>>(Wih0, wih0p, GAT, INP);
  pack_w<<<(GAT/16 * HID/32 * 32 + 255) / 256, 256, 0, stream>>>(Whh0, whh0p, GAT, HID);
  pack_w<<<(GAT/16 * HID/32 * 32 + 255) / 256, 256, 0, stream>>>(Wih1, wih1p, GAT, HID);
  pack_w<<<(GAT/16 * HID/32 * 32 + 255) / 256, 256, 0, stream>>>(Whh1, whh1p, GAT, HID);
  bias_sum<<<(GAT + 255) / 256, 256, 0, stream>>>(bih0, bhh0, bs0, GAT);
  bias_sum<<<(GAT + 255) / 256, 256, 0, stream>>>(bih1, bhh1, bs1, GAT);
  {
    size_t nx = (size_t)BATCH * SEQT * INP;
    f32_to_bf16_k<<<(unsigned)((nx + 255) / 256), 256, 0, stream>>>(x, x_bf, nx);
  }

  const int MT = BATCH * SEQT / 16;              // 8192 row tiles
  const int gemm_blocks = MT * (GAT / 16) / 8;   // 8 waves/block

  // --- layer 0 ---
  gemm_xw<<<gemm_blocks, 256, 0, stream>>>(x_bf, wih0p, bs0, xw_bf, INP);
  hipMemsetAsync(hb0, 0, (size_t)BATCH * HID * 2, stream);
  hipMemsetAsync(hb1, 0, (size_t)BATCH * HID * 2, stream);
  hipMemsetAsync(bar, 0, 256, stream);
  lstm_recur<<<32, 256, 0, stream>>>(xw_bf, whh0p, hb0, hb1, h1_bf, nullptr, bar);

  // --- layer 1 ---
  gemm_xw<<<gemm_blocks, 256, 0, stream>>>(h1_bf, wih1p, bs1, xw_bf, HID);
  hipMemsetAsync(hb0, 0, (size_t)BATCH * HID * 2, stream);
  hipMemsetAsync(hb1, 0, (size_t)BATCH * HID * 2, stream);
  hipMemsetAsync(bar, 0, 256, stream);
  lstm_recur<<<32, 256, 0, stream>>>(xw_bf, whh1p, hb0, hb1, nullptr, hlast, bar);

  // --- final FC ---
  fc_kernel<<<1, 256, 0, stream>>>(hlast, fcw, fcb, out);
}
